// FFLSTM_88450556494216
// MI455X (gfx1250) — matrix operand order repaired and checked by reading, not compiled, not run on hardware
//
#include <hip/hip_runtime.h>
#include <hip/hip_bf16.h>

// ---------------------------------------------------------------------------
// Problem constants (from reference): T=128, B=256, F=512, H=512, L=2
// ---------------------------------------------------------------------------
#define T_STEPS  128
#define BATCH    256
#define FEAT     512
#define HID      512
#define NBLOCKS  64            // 2 blocks per H-tile (32 H-tiles of 16 cols)
#define NTHREADS 256           // 8 waves/block = 8 batch tiles of 16 rows

// LDS regions: per block, weights for its 16-column H-tile, all 4 gates.
// Each cell matrix tile: 16 ksteps x 4 gates x 1KB fragment = 64KB.
#define OFF_W10  0             // w_ih0 tile
#define OFF_W20  65536         // w_hh0 tile
#define OFF_W11  131072        // w_ih1 tile
#define OFF_W21  196608        // w_hh1 tile
#define OFF_WO   262144        // w_out tile: 16 ksteps x 1KB = 16KB
#define LDS_BYTES 278528       // 272KB <= 320KB WGP LDS  [CDNA5]

typedef __attribute__((ext_vector_type(16))) __bf16 bf16x16;
typedef __attribute__((ext_vector_type(8)))  float  f32x8;

union Frag {
    uint4        q[2];
    unsigned int u[8];
    bf16x16      v;
};

#define WMMA_BF16(a, b, c) \
    __builtin_amdgcn_wmma_f32_16x16x32_bf16(false, (a), false, (b), (short)0, (c), false, false)

// ---------------------------------------------------------------------------
// A-fragment (16x32 MxK bf16) from global, per CDNA5 wave32 layout:
// lane&15 = row M; lanes 0-15 hold K 0-7 & 16-23, lanes 16-31 hold K 8-15 &
// 24-31 (2 bf16/VGPR, contiguous in K -> two 16B global loads).
// ---------------------------------------------------------------------------
__device__ __forceinline__ bf16x16 load_frag_a(const __hip_bfloat16* base,
                                               int ld, int m0, int k0, int lane) {
    const int m  = m0 + (lane & 15);
    const int kb = (lane >> 4) << 3;                     // 0 or 8
    const unsigned short* row = (const unsigned short*)(base + (size_t)m * ld + k0 + kb);
    Frag f;
    f.q[0] = *(const uint4*)(row);
    f.q[1] = *(const uint4*)(row + 16);
    return f.v;
}

// B-fragment (32x16 KxN) gathered from W[N,K] row-major in global (staging only).
__device__ __forceinline__ Frag load_frag_b_raw(const __hip_bfloat16* W,
                                                int ld, int n0, int k0, int lane) {
    const int n  = n0 + (lane & 15);
    const int kb = (lane >> 4) << 3;                     // 0 or 8
    const unsigned short* row = (const unsigned short*)(W + (size_t)n * ld + k0 + kb);
    Frag f;
    f.q[0] = *(const uint4*)(row);
    f.q[1] = *(const uint4*)(row + 16);
    return f;
}

// ---------------------------------------------------------------------------
// LDS fragment store/load. Fragment fi occupies 1KB; lane l's dword-pair j
// lives at fi*1024 + (j*32 + l)*8  -> ds_load_b64 banks {2l,2l+1}: conflict-
// free across the 64 LDS banks. Staged and read with identical lane mapping.
// ---------------------------------------------------------------------------
__device__ __forceinline__ void lds_store_frag(char* region, int fi, int lane, Frag f) {
    unsigned long long* p = (unsigned long long*)(region + fi * 1024 + lane * 8);
#pragma unroll
    for (int j = 0; j < 4; ++j)
        p[j * 32] = (unsigned long long)f.u[2 * j] |
                    ((unsigned long long)f.u[2 * j + 1] << 32);
}

__device__ __forceinline__ bf16x16 lds_load_frag(const char* region, int fi, int lane) {
    const unsigned long long* p =
        (const unsigned long long*)(region + fi * 1024 + lane * 8);
    Frag f;
#pragma unroll
    for (int j = 0; j < 4; ++j) {
        const unsigned long long d = p[j * 32];
        f.u[2 * j]     = (unsigned int)d;
        f.u[2 * j + 1] = (unsigned int)(d >> 32);
    }
    return f.v;
}

// Stage one weight tile into LDS: ngate*16 fragments, wave wv stages its share.
__device__ __forceinline__ void stage_region(char* region, const __hip_bfloat16* W,
                                             int ld, int ncol0, int ngate,
                                             int wv, int lane) {
    const int per_wave = (16 * ngate) / 8;
    for (int fi = wv * per_wave; fi < (wv + 1) * per_wave; ++fi) {
        const int ks = fi / ngate, g = fi % ngate;
        lds_store_frag(region, fi, lane,
                       load_frag_b_raw(W, ld, g * HID + ncol0, ks * 32, lane));
    }
}

__device__ __forceinline__ float sigmoidf_(float x) { return 1.0f / (1.0f + __expf(-x)); }

// ---------------------------------------------------------------------------
// Device-wide sense/generation barrier (64 co-resident blocks).
// ---------------------------------------------------------------------------
__device__ __forceinline__ void grid_barrier(unsigned int* cnt, unsigned int* gen) {
    __syncthreads();
    if (threadIdx.x == 0) {
        __threadfence();
        unsigned int g = __hip_atomic_load(gen, __ATOMIC_RELAXED, __HIP_MEMORY_SCOPE_AGENT);
        unsigned int a = __hip_atomic_fetch_add(cnt, 1u, __ATOMIC_ACQ_REL, __HIP_MEMORY_SCOPE_AGENT);
        if (a == NBLOCKS - 1) {
            __hip_atomic_store(cnt, 0u, __ATOMIC_RELAXED, __HIP_MEMORY_SCOPE_AGENT);
            __hip_atomic_fetch_add(gen, 1u, __ATOMIC_ACQ_REL, __HIP_MEMORY_SCOPE_AGENT);
        } else {
            while (__hip_atomic_load(gen, __ATOMIC_ACQUIRE, __HIP_MEMORY_SCOPE_AGENT) == g)
                __builtin_amdgcn_s_sleep(1);
        }
        __threadfence();
    }
    __syncthreads();
}

// ---------------------------------------------------------------------------
// Fused LSTM cell: gates = A1@W1^T + A2@W2^T (+bias); i,f,g,o -> h,c.
// B-fragments from LDS (weights-stationary); A prefetched one step ahead.
// NOTE: no __restrict__ on state buffers (they alias across stages).
// ---------------------------------------------------------------------------
__device__ __forceinline__ void stage_cell(
    int m0, int n0, int lane,
    const __hip_bfloat16* A1, const char* ldsW1,
    const __hip_bfloat16* A2, const char* ldsW2,
    const float* bias, float* c, __hip_bfloat16* h_out,
    const float* resid_in, __hip_bfloat16* sum_out)
{
    f32x8 acc[4] = {};                       // i, f, g, o

    bf16x16 a_cur = load_frag_a(A1, FEAT, m0, 0, lane);
#pragma unroll
    for (int ks = 0; ks < 16; ++ks) {        // part 1: A1 @ W1^T
        const bf16x16 a_nxt = (ks < 15)
            ? load_frag_a(A1, FEAT, m0, (ks + 1) * 32, lane)
            : load_frag_a(A2, HID, m0, 0, lane);   // bridge into part 2
#pragma unroll
        for (int g = 0; g < 4; ++g) {
            const bf16x16 b = lds_load_frag(ldsW1, ks * 4 + g, lane);
            acc[g] = WMMA_BF16(a_cur, b, acc[g]);
        }
        a_cur = a_nxt;
    }
#pragma unroll
    for (int ks = 0; ks < 16; ++ks) {        // part 2: A2 @ W2^T
        bf16x16 a_nxt = a_cur;
        if (ks < 15) a_nxt = load_frag_a(A2, HID, m0, (ks + 1) * 32, lane);
#pragma unroll
        for (int g = 0; g < 4; ++g) {
            const bf16x16 b = lds_load_frag(ldsW2, ks * 4 + g, lane);
            acc[g] = WMMA_BF16(a_cur, b, acc[g]);
        }
        a_cur = a_nxt;
    }

    // Epilogue: C/D layout -> lane&15 = col, VGPR v = row (v + 8*(lane>=16)).
    const int col   = n0 + (lane & 15);
    const int rbase = m0 + ((lane >> 4) << 3);
    float bi = 0.f, bff = 0.f, bg = 0.f, bo = 0.f;
    if (bias) {
        bi  = bias[0 * HID + col];
        bff = bias[1 * HID + col];
        bg  = bias[2 * HID + col];
        bo  = bias[3 * HID + col];
    }
#pragma unroll
    for (int v = 0; v < 8; ++v) {
        const int   idx = (rbase + v) * HID + col;
        const float gi = acc[0][v] + bi;
        const float gf = acc[1][v] + bff;
        const float gg = acc[2][v] + bg;
        const float go = acc[3][v] + bo;
        const float cp = c[idx];
        const float cn = sigmoidf_(gf) * cp + sigmoidf_(gi) * tanhf(gg);
        const float hn = sigmoidf_(go) * tanhf(cn);
        c[idx]     = cn;
        h_out[idx] = __float2bfloat16(hn);
        if (sum_out) {
            const float s = hn + (resid_in ? resid_in[idx] : 0.0f);
            sum_out[idx] = __float2bfloat16(s);
        }
    }
}

// ---------------------------------------------------------------------------
// Projection: out = A @ W^T + b. Wave = one 16x16 tile; W-fragments from LDS.
// ---------------------------------------------------------------------------
__device__ __forceinline__ void stage_proj(
    int m0, int n0, int lane,
    const __hip_bfloat16* A, const char* ldsWO, const float* bias,
    float* out_f32, __hip_bfloat16* out_bf)
{
    f32x8 acc = {};
    bf16x16 a_cur = load_frag_a(A, HID, m0, 0, lane);
#pragma unroll
    for (int ks = 0; ks < 16; ++ks) {
        bf16x16 a_nxt = a_cur;
        if (ks < 15) a_nxt = load_frag_a(A, HID, m0, (ks + 1) * 32, lane);
        const bf16x16 b = lds_load_frag(ldsWO, ks, lane);
        acc = WMMA_BF16(a_cur, b, acc);
        a_cur = a_nxt;
    }

    const int col   = n0 + (lane & 15);
    const int rbase = m0 + ((lane >> 4) << 3);
    const float bb = bias[col];
#pragma unroll
    for (int v = 0; v < 8; ++v) {
        const float val = acc[v] + bb;
        const int   idx = (rbase + v) * HID + col;
        if (out_f32) out_f32[idx] = val;
        if (out_bf)  out_bf[idx]  = __float2bfloat16(val);
    }
}

// ---------------------------------------------------------------------------
// Persistent kernel: weights staged ONCE into LDS (272KB, weights-stationary),
// then the whole T=128 recurrence runs in one launch with grid barriers.
// Block -> H-tile nb = blockIdx>>1 (16 cols, all 4 gates); wave -> batch tile.
// ---------------------------------------------------------------------------
__global__ void __launch_bounds__(NTHREADS)
lstm_persistent(const __hip_bfloat16* x_bf,
                const __hip_bfloat16* wih0, const __hip_bfloat16* whh0,
                const __hip_bfloat16* wih1, const __hip_bfloat16* whh1,
                const __hip_bfloat16* wout,
                const float* b1, const float* b_out,
                __hip_bfloat16* h0a, __hip_bfloat16* h0b,
                __hip_bfloat16* h1a, __hip_bfloat16* h1b,
                float* c0, float* c1,
                float* f_f32, __hip_bfloat16* f_bf, __hip_bfloat16* hf_bf,
                float* out,
                unsigned int* bar_cnt, unsigned int* bar_gen)
{
    extern __shared__ char smem[];

    const int wv    = threadIdx.x >> 5;          // 0..7
    const int lane  = threadIdx.x & 31;
    const int nb    = blockIdx.x >> 1;           // 0..31 H-tile
    const int mhalf = blockIdx.x & 1;            // 0..1
    const int mtile = mhalf * 8 + wv;            // 0..15 batch tile
    const int m0 = mtile * 16, n0 = nb * 16;

    // ---- one-time weight staging into LDS ----
    stage_region(smem + OFF_W10, wih0, FEAT, n0, 4, wv, lane);
    stage_region(smem + OFF_W20, whh0, HID,  n0, 4, wv, lane);
    stage_region(smem + OFF_W11, wih1, HID,  n0, 4, wv, lane);
    stage_region(smem + OFF_W21, whh1, HID,  n0, 4, wv, lane);
    stage_region(smem + OFF_WO,  wout, HID,  n0, 1, wv, lane);
    __syncthreads();

    __hip_bfloat16* h0p[2] = { h0a, h0b };
    __hip_bfloat16* h1p[2] = { h1a, h1b };

    for (int t = 0; t < T_STEPS; ++t) {
        // Launder a zero offset so the compiler cannot hoist the (loop-
        // invariant) LDS fragment loads out of the t-loop into registers
        // (which caused catastrophic scratch spills). Keeps LDS addrspace.
        int zoff = 0;
        asm volatile("" : "+v"(zoff));
        const char* pW10 = smem + OFF_W10 + zoff;
        const char* pW20 = smem + OFF_W20 + zoff;
        const char* pW11 = smem + OFF_W11 + zoff;
        const char* pW21 = smem + OFF_W21 + zoff;
        const char* pWO  = smem + OFF_WO  + zoff;

        const int cur = t & 1, nxt = cur ^ 1;
        const __hip_bfloat16* xt = x_bf + (size_t)t * BATCH * FEAT;

        // layer 0 cell (no bias)
        stage_cell(m0, n0, lane, xt, pW10, h0p[cur], pW20, nullptr,
                   c0, h0p[nxt], nullptr, nullptr);
        grid_barrier(bar_cnt, bar_gen);

        // f = h0 @ w_out^T + b_out  (f32 for residual, bf16 for next GEMM)
        stage_proj(m0, n0, lane, h0p[nxt], pWO, b_out, f_f32, f_bf);
        grid_barrier(bar_cnt, bar_gen);

        // layer 1 cell (+bias, +residual sum)
        stage_cell(m0, n0, lane, f_bf, pW11, h1p[cur], pW21, b1,
                   c1, h1p[nxt], f_f32, hf_bf);
        grid_barrier(bar_cnt, bar_gen);

        // out_t = (h1 + f) @ w_out^T + b_out
        stage_proj(m0, n0, lane, hf_bf, pWO, b_out,
                   out + (size_t)t * BATCH * HID, nullptr);
        grid_barrier(bar_cnt, bar_gen);
    }
}

// ---------------------------------------------------------------------------
// Prep kernels
// ---------------------------------------------------------------------------
__global__ void f32_to_bf16_kernel(const float* __restrict__ in,
                                   __hip_bfloat16* __restrict__ out, int n) {
    for (int i = blockIdx.x * blockDim.x + threadIdx.x; i < n;
         i += gridDim.x * blockDim.x)
        out[i] = __float2bfloat16(in[i]);
}

__global__ void add_vec_kernel(const float* __restrict__ a,
                               const float* __restrict__ b,
                               float* __restrict__ o, int n) {
    for (int i = blockIdx.x * blockDim.x + threadIdx.x; i < n;
         i += gridDim.x * blockDim.x)
        o[i] = a[i] + b[i];
}

__global__ void zero_u32_kernel(unsigned int* __restrict__ p, int n) {
    for (int i = blockIdx.x * blockDim.x + threadIdx.x; i < n;
         i += gridDim.x * blockDim.x)
        p[i] = 0u;
}

// ---------------------------------------------------------------------------
// Host-side orchestration
// ---------------------------------------------------------------------------
extern "C" void kernel_launch(void* const* d_in, const int* in_sizes, int n_in,
                              void* d_out, int out_size, void* d_ws, size_t ws_size,
                              hipStream_t stream) {
    (void)in_sizes; (void)n_in; (void)out_size; (void)ws_size;

    const float* x_f32   = (const float*)d_in[0];
    const float* w_ih0_f = (const float*)d_in[1];
    const float* w_hh0_f = (const float*)d_in[2];
    const float* w_ih1_f = (const float*)d_in[3];
    const float* w_hh1_f = (const float*)d_in[4];
    const float* b_ih1   = (const float*)d_in[5];
    const float* b_hh1   = (const float*)d_in[6];
    const float* w_out_f = (const float*)d_in[7];
    const float* b_out   = (const float*)d_in[8];
    float* out = (float*)d_out;

    char* wp = (char*)d_ws;
    auto alloc = [&](size_t bytes) -> void* {
        void* p = (void*)wp;
        wp += (bytes + 255) & ~(size_t)255;
        return p;
    };
    const size_t NXB = (size_t)T_STEPS * BATCH * FEAT;
    const size_t NW  = (size_t)4 * HID * FEAT;
    const size_t NWO = (size_t)HID * HID;
    const size_t NBH = (size_t)BATCH * HID;

    __hip_bfloat16* x_bf    = (__hip_bfloat16*)alloc(NXB * 2);
    __hip_bfloat16* wih0_bf = (__hip_bfloat16*)alloc(NW * 2);
    __hip_bfloat16* whh0_bf = (__hip_bfloat16*)alloc(NW * 2);
    __hip_bfloat16* wih1_bf = (__hip_bfloat16*)alloc(NW * 2);
    __hip_bfloat16* whh1_bf = (__hip_bfloat16*)alloc(NW * 2);
    __hip_bfloat16* wout_bf = (__hip_bfloat16*)alloc(NWO * 2);
    float*          b1      = (float*)alloc(4 * HID * 4);
    __hip_bfloat16* h0a     = (__hip_bfloat16*)alloc(NBH * 2);
    __hip_bfloat16* h0b     = (__hip_bfloat16*)alloc(NBH * 2);
    __hip_bfloat16* h1a     = (__hip_bfloat16*)alloc(NBH * 2);
    __hip_bfloat16* h1b     = (__hip_bfloat16*)alloc(NBH * 2);
    float*          c0      = (float*)alloc(NBH * 4);
    float*          c1      = (float*)alloc(NBH * 4);
    float*          f_f32   = (float*)alloc(NBH * 4);
    __hip_bfloat16* f_bf    = (__hip_bfloat16*)alloc(NBH * 2);
    __hip_bfloat16* hf_bf   = (__hip_bfloat16*)alloc(NBH * 2);
    unsigned int*   bar     = (unsigned int*)alloc(2 * sizeof(unsigned int));

    f32_to_bf16_kernel<<<2048, 256, 0, stream>>>(x_f32, x_bf, (int)NXB);
    f32_to_bf16_kernel<<<512,  256, 0, stream>>>(w_ih0_f, wih0_bf, (int)NW);
    f32_to_bf16_kernel<<<512,  256, 0, stream>>>(w_hh0_f, whh0_bf, (int)NW);
    f32_to_bf16_kernel<<<512,  256, 0, stream>>>(w_ih1_f, wih1_bf, (int)NW);
    f32_to_bf16_kernel<<<512,  256, 0, stream>>>(w_hh1_f, whh1_bf, (int)NW);
    f32_to_bf16_kernel<<<256,  256, 0, stream>>>(w_out_f, wout_bf, (int)NWO);
    add_vec_kernel<<<8, 256, 0, stream>>>(b_ih1, b_hh1, b1, 4 * HID);
    zero_u32_kernel<<<128, 256, 0, stream>>>((unsigned int*)h0a, (int)(NBH / 2));
    zero_u32_kernel<<<128, 256, 0, stream>>>((unsigned int*)h1a, (int)(NBH / 2));
    zero_u32_kernel<<<256, 256, 0, stream>>>((unsigned int*)c0, (int)NBH);
    zero_u32_kernel<<<256, 256, 0, stream>>>((unsigned int*)c1, (int)NBH);
    zero_u32_kernel<<<1, 64, 0, stream>>>(bar, 2);

    lstm_persistent<<<NBLOCKS, NTHREADS, LDS_BYTES, stream>>>(
        x_bf, wih0_bf, whh0_bf, wih1_bf, whh1_bf, wout_bf, b1, b_out,
        h0a, h0b, h1a, h1b, c0, c1, f_f32, f_bf, hf_bf, out,
        bar + 0, bar + 1);
}
